// Decoder_58600533787128
// MI455X (gfx1250) — matrix operand order repaired and checked by reading, not compiled, not run on hardware
//
#include <hip/hip_runtime.h>
#include <hip/hip_bf16.h>

typedef _Float16 h8  __attribute__((ext_vector_type(8)));
typedef _Float16 h16 __attribute__((ext_vector_type(16)));
typedef float    f8  __attribute__((ext_vector_type(8)));

#define WMMA_F16(a, b, c) \
  __builtin_amdgcn_wmma_f32_16x16x32_f16(false, (a), false, (b), (short)0, (c), false, false)

// Problem constants
constexpr int Bc = 64, Nc = 64, NIN = 64, NHID = 256, NOUT = 64, Tc = 4, Ec = 4032;

// Workspace byte offsets
constexpr size_t OFF_AGG  = 0;                       // [B][N][64] f32   (1 MB)
constexpr size_t OFF_X16  = 1048576;                 // [B][N][64] f16
constexpr size_t OFF_W1T  = OFF_X16 + 524288;        // [T][256][128] f16
constexpr size_t OFF_W2T  = OFF_W1T + 262144;        // [T][64][256]  f16
constexpr size_t OFF_OW1T = OFF_W2T + 131072;        // [256][128] f16
constexpr size_t OFF_OW2T = OFF_OW1T + 65536;        // [256][256] f16
constexpr size_t OFF_OW3T = OFF_OW2T + 131072;       // [64][256]  f16

static __device__ __forceinline__ h16 cat8(h8 lo, h8 hi) {
  return __builtin_shufflevector(lo, hi, 0, 1, 2, 3, 4, 5, 6, 7,
                                         8, 9, 10, 11, 12, 13, 14, 15);
}

// A fragment (16x32 f16): lanes 0-15 row M=lane get K[kb+0..16);
// lanes 16-31 same row, K[kb+16..32). rowptr already includes row*ldk.
static __device__ __forceinline__ h16 afrag_ptr(const _Float16* rowptr, int kb) {
  const int lane = threadIdx.x & 31;
  const int s = (lane & 16) ? 16 : 0;
  const _Float16* p = rowptr + kb + s;
  h8 lo = *(const h8*)p;
  h8 hi = *(const h8*)(p + 8);
  return cat8(lo, hi);
}

// ---------------------------------------------------------------------------
// prep: fp32 -> fp16 conversions + weight transposes (n-major for B-fragments)
// ---------------------------------------------------------------------------
__global__ __launch_bounds__(256) void prep_kernel(
    const float* __restrict__ x, const float* __restrict__ w1,
    const float* __restrict__ w2, const float* __restrict__ ow1,
    const float* __restrict__ ow2, const float* __restrict__ ow3,
    _Float16* __restrict__ x16, _Float16* __restrict__ w1t,
    _Float16* __restrict__ w2t, _Float16* __restrict__ ow1t,
    _Float16* __restrict__ ow2t, _Float16* __restrict__ ow3t) {
  const int i = blockIdx.x * 256 + threadIdx.x;
  if (i < 262144) x16[i] = (_Float16)x[i];
  if (i < 131072) {  // w1 [T][128][256] -> w1t [T][256][128]
    int t = i >> 15, r = i & 32767, n = r >> 7, k = r & 127;
    w1t[i] = (_Float16)w1[(t << 15) + k * 256 + n];
  }
  if (i < 65536) {   // w2 [T][256][64] -> w2t [T][64][256]
    int t = i >> 14, r = i & 16383, n = r >> 8, k = r & 255;
    w2t[i] = (_Float16)w2[(t << 14) + k * 64 + n];
  }
  if (i < 32768) {   // ow1 [128][256] -> ow1t [256][128]
    int n = i >> 7, k = i & 127;
    ow1t[i] = (_Float16)ow1[k * 256 + n];
  }
  if (i < 65536) {   // ow2 [256][256] -> ow2t [256][256]
    int n = i >> 8, k = i & 255;
    ow2t[i] = (_Float16)ow2[k * 256 + n];
  }
  if (i < 16384) {   // ow3 [256][64] -> ow3t [64][256]
    int n = i >> 8, k = i & 255;
    ow3t[i] = (_Float16)ow3[k * 64 + n];
  }
}

// ---------------------------------------------------------------------------
// edge kernel: per (receiver-block, batch). Fused gather -> MLP(x4 types)
// -> rel_type weighting -> receiver aggregation.
// Uncapped register budget: >256 VGPRs means 1 WG/WGP (2 waves/SIMD) either
// way, and the extra registers buy spill-free deep pipelining (16 in-flight
// A-fragments, all B-fragments persistent) at the cost of co-issued
// s_set_vgpr_msb SALU.
// ---------------------------------------------------------------------------
__global__ __launch_bounds__(256) void edge_kernel(
    const _Float16* __restrict__ x16, const float* __restrict__ rel_type,
    const _Float16* __restrict__ w1t, const float* __restrict__ b1,
    const _Float16* __restrict__ w2t, const float* __restrict__ b2,
    float* __restrict__ agg) {
  __shared__ __align__(16) unsigned char smem[8192 + 65536];
  _Float16* xb = (_Float16*)smem;            // [64][64] f16 node features
  _Float16* hs = (_Float16*)(smem + 8192);   // [128][256] f16 hidden chunk
  float*    red = (float*)(smem + 8192);     // [8][16][64] f32 (aliases hs)

  const int rblk = blockIdx.x;       // 0..3  (receiver block)
  const int b    = blockIdx.y;       // 0..63 (batch)
  const int R0   = rblk * 16;
  const int tid  = threadIdx.x;
  const int wave = tid >> 5;
  const int lane = tid & 31;
  const int l15  = lane & 15;
  const int hi   = (lane & 16) ? 1 : 0;
  const int hib  = hi ? 16 : 0;  // B-fragment K offset
  const int hir  = hi ? 8 : 0;   // C row offset

  // Stage x[b] (64x64 f16 = 8 KB) into LDS
  {
    const uint4* src = (const uint4*)(x16 + (size_t)b * (Nc * NIN));
    uint4* dst = (uint4*)xb;
    dst[tid] = src[tid];
    dst[tid + 256] = src[tid + 256];
  }
  __syncthreads();

  // Receiver-half A fragments: invariant across chunks AND types. Load once.
  h16 rf[2];
  rf[0] = afrag_ptr(xb + (R0 + l15) * 64, 0);
  rf[1] = afrag_ptr(xb + (R0 + l15) * 64, 32);

  // Persistent per-wave aggregation registers: [4 n-tiles][8] over 16 rr x 64 nf
  f8 aggr[4];
#pragma unroll
  for (int n = 0; n < 4; ++n)
#pragma unroll
    for (int v = 0; v < 8; ++v) aggr[n][v] = 0.0f;

  for (int t = 0; t < Tc; ++t) {
    const _Float16* W1 = w1t + (size_t)t * (256 * 128);
    const _Float16* W2 = w2t + (size_t)t * (64 * 256);
    const float bia0 = b1[t * 256 + wave * 32 + l15];
    const float bia1 = b1[t * 256 + wave * 32 + 16 + l15];
    float bia2[4];
#pragma unroll
    for (int n = 0; n < 4; ++n) bia2[n] = b2[t * 64 + n * 16 + l15];

    // Layer-1 B fragments for this type: held in registers across all chunks.
    h16 bfa[4], bfb[4];
#pragma unroll
    for (int k = 0; k < 4; ++k) {
      bfa[k] = *(const h16*)(W1 + (wave * 32 + l15) * 128 + k * 32 + hib);
      bfb[k] = *(const h16*)(W1 + (wave * 32 + 16 + l15) * 128 + k * 32 + hib);
    }

    for (int chunk = 0; chunk < 8; ++chunk) {
      // ---- layer 1 in two m-passes (4 M-tiles live at a time) ----
#pragma unroll
      for (int half = 0; half < 2; ++half) {
        const int mb = half * 4;
        f8 c1[4][2];
#pragma unroll
        for (int m = 0; m < 4; ++m)
#pragma unroll
          for (int j = 0; j < 2; ++j)
#pragma unroll
            for (int v = 0; v < 8; ++v) c1[m][j][v] = 0.0f;

        // Sender half (K 0..63): per-m broadcast rows; batch loads, then WMMAs
#pragma unroll
        for (int k = 0; k < 2; ++k) {
          h16 af[4];
#pragma unroll
          for (int mm = 0; mm < 4; ++mm)
            af[mm] = afrag_ptr(xb + (chunk * 8 + mb + mm) * 64, k * 32);
#pragma unroll
          for (int mm = 0; mm < 4; ++mm) {
            c1[mm][0] = WMMA_F16(af[mm], bfa[k], c1[mm][0]);
            c1[mm][1] = WMMA_F16(af[mm], bfb[k], c1[mm][1]);
          }
        }
        // Receiver half (K 64..127): shared fragment held in registers
#pragma unroll
        for (int k = 2; k < 4; ++k) {
#pragma unroll
          for (int mm = 0; mm < 4; ++mm) {
            c1[mm][0] = WMMA_F16(rf[k - 2], bfa[k], c1[mm][0]);
            c1[mm][1] = WMMA_F16(rf[k - 2], bfb[k], c1[mm][1]);
          }
        }
        if (half == 0) __syncthreads();  // prior chunk's layer-2 hs reads done
        // H = relu(c1 + b1) -> hs rows [mb*16, mb*16+64)
#pragma unroll
        for (int mm = 0; mm < 4; ++mm)
#pragma unroll
          for (int j = 0; j < 2; ++j) {
            const int col = wave * 32 + j * 16 + l15;
            const float bi = j ? bia1 : bia0;
#pragma unroll
            for (int v = 0; v < 8; ++v) {
              const int row = (mb + mm) * 16 + v + hir;
              float val = c1[mm][j][v] + bi;
              hs[row * 256 + col] = (_Float16)(val > 0.0f ? val : 0.0f);
            }
          }
      }
      __syncthreads();  // hs fully written

      // ---- layer 2: wave w owns m-tile w (sender i = chunk*8 + w) ----
      f8 c2[4];
#pragma unroll
      for (int n = 0; n < 4; ++n)
#pragma unroll
        for (int v = 0; v < 8; ++v) c2[n][v] = 0.0f;
#pragma unroll
      for (int k = 0; k < 8; ++k) {
        h16 af = afrag_ptr(hs + (wave * 16 + l15) * 256, k * 32);
#pragma unroll
        for (int n = 0; n < 4; ++n) {
          h16 bf = *(const h16*)(W2 + (n * 16 + l15) * 256 + k * 32 + hib);
          c2[n] = WMMA_F16(af, bf, c2[n]);
        }
      }

      // ---- epilogue: relu(c2+b2) * rel_type, accumulate over senders ----
      // Branchless (no EXEC divergence): clamped index + select.
      const int snd = chunk * 8 + wave;
      float rw[8];
#pragma unroll
      for (int v = 0; v < 8; ++v) {
        const int r = R0 + v + hir;
        const bool self = (r == snd);
        const int eidx = snd * 63 + r - (r > snd ? 1 : 0);
        const int esafe = self ? 0 : eidx;
        const float rv = rel_type[((size_t)b * Ec + esafe) * Tc + t];
        rw[v] = self ? 0.0f : rv;
      }
#pragma unroll
      for (int n = 0; n < 4; ++n)
#pragma unroll
        for (int v = 0; v < 8; ++v) {
          float m2 = c2[n][v] + bia2[n];
          m2 = m2 > 0.0f ? m2 : 0.0f;
          aggr[n][v] += m2 * rw[v];
        }
    }  // chunk
  }    // t

  __syncthreads();  // all hs reads finished; red may alias hs now
  // Per-wave partials -> LDS
#pragma unroll
  for (int n = 0; n < 4; ++n)
#pragma unroll
    for (int v = 0; v < 8; ++v) {
      const int rr = v + hir;
      red[(wave * 16 + rr) * 64 + n * 16 + l15] = aggr[n][v];
    }
  __syncthreads();
  // Deterministic fixed-order reduction over the 8 waves
#pragma unroll
  for (int q = 0; q < 4; ++q) {
    const int idx = q * 256 + tid;      // 0..1023 over [16 rr][64 nf]
    const int rr = idx >> 6, nf = idx & 63;
    float s = 0.0f;
#pragma unroll
    for (int wv = 0; wv < 8; ++wv) s += red[(wv * 16 + rr) * 64 + nf];
    agg[((size_t)b * Nc + R0 + rr) * 64 + nf] = s;
  }
}

// ---------------------------------------------------------------------------
// node kernel: aug=[x|agg] -> relu(.@ow1+ob1) -> relu(.@ow2+ob2) -> .@ow3+ob3
// ---------------------------------------------------------------------------
__global__ __launch_bounds__(256) void node_kernel(
    const _Float16* __restrict__ x16, const float* __restrict__ aggf,
    const _Float16* __restrict__ ow1t, const float* __restrict__ ob1,
    const _Float16* __restrict__ ow2t, const float* __restrict__ ob2,
    const _Float16* __restrict__ ow3t, const float* __restrict__ ob3,
    float* __restrict__ out) {
  __shared__ __align__(16) unsigned char smem[131072];
  _Float16* h1  = (_Float16*)smem;            // [128][256]
  _Float16* aug = (_Float16*)(smem + 65536);  // [128][128] (dead after L1)
  _Float16* h2  = (_Float16*)(smem + 65536);  // [128][256] (aliases aug)

  const int tid  = threadIdx.x;
  const int wave = tid >> 5;
  const int lane = tid & 31;
  const int l15  = lane & 15;
  const int hi   = (lane & 16) ? 1 : 0;
  const int hib  = hi ? 16 : 0;
  const int hir  = hi ? 8 : 0;
  const int row0 = blockIdx.x * 128;  // global row = b*64 + node

  // Stage aug rows: x (f16 copy) | agg (f32 -> f16)
  {
    const int row = tid >> 1, half = tid & 1;
    _Float16* d = aug + row * 128 + half * 64;
    if (half == 0) {
      const uint4* s = (const uint4*)(x16 + (size_t)(row0 + row) * 64);
      uint4* dv = (uint4*)d;
#pragma unroll
      for (int q = 0; q < 8; ++q) dv[q] = s[q];
    } else {
      const float* s = aggf + (size_t)(row0 + row) * 64;
#pragma unroll
      for (int q = 0; q < 64; ++q) d[q] = (_Float16)s[q];
    }
  }
  __syncthreads();

  // ---- L1: [128][128] @ ow1t -> h1[128][256], relu + ob1 (two m-passes) ----
  {
    const float bi0 = ob1[wave * 32 + l15];
    const float bi1 = ob1[wave * 32 + 16 + l15];
#pragma unroll
    for (int half = 0; half < 2; ++half) {
      const int mb = half * 4;
      f8 c[4][2];
#pragma unroll
      for (int m = 0; m < 4; ++m)
#pragma unroll
        for (int j = 0; j < 2; ++j)
#pragma unroll
          for (int v = 0; v < 8; ++v) c[m][j][v] = 0.0f;
#pragma unroll
      for (int k = 0; k < 4; ++k) {
        h16 bf0 = *(const h16*)(ow1t + (wave * 32 + l15) * 128 + k * 32 + hib);
        h16 bf1 = *(const h16*)(ow1t + (wave * 32 + 16 + l15) * 128 + k * 32 + hib);
        h16 af[4];
#pragma unroll
        for (int mm = 0; mm < 4; ++mm)
          af[mm] = afrag_ptr(aug + ((mb + mm) * 16 + l15) * 128, k * 32);
#pragma unroll
        for (int mm = 0; mm < 4; ++mm) {
          c[mm][0] = WMMA_F16(af[mm], bf0, c[mm][0]);
          c[mm][1] = WMMA_F16(af[mm], bf1, c[mm][1]);
        }
      }
#pragma unroll
      for (int mm = 0; mm < 4; ++mm)
#pragma unroll
        for (int j = 0; j < 2; ++j) {
          const int col = wave * 32 + j * 16 + l15;
          const float bi = j ? bi1 : bi0;
#pragma unroll
          for (int v = 0; v < 8; ++v) {
            float val = c[mm][j][v] + bi;
            h1[((mb + mm) * 16 + v + hir) * 256 + col] =
                (_Float16)(val > 0.0f ? val : 0.0f);
          }
        }
    }
  }
  __syncthreads();  // h1 ready; aug reads complete (safe to overwrite with h2)

  // ---- L2: [128][256] @ ow2t -> h2[128][256], relu + ob2 (two m-passes) ----
  {
    const float bi0 = ob2[wave * 32 + l15];
    const float bi1 = ob2[wave * 32 + 16 + l15];
#pragma unroll
    for (int half = 0; half < 2; ++half) {
      const int mb = half * 4;
      f8 c[4][2];
#pragma unroll
      for (int m = 0; m < 4; ++m)
#pragma unroll
        for (int j = 0; j < 2; ++j)
#pragma unroll
          for (int v = 0; v < 8; ++v) c[m][j][v] = 0.0f;
#pragma unroll
      for (int k = 0; k < 8; ++k) {
        h16 bf0 = *(const h16*)(ow2t + (wave * 32 + l15) * 256 + k * 32 + hib);
        h16 bf1 = *(const h16*)(ow2t + (wave * 32 + 16 + l15) * 256 + k * 32 + hib);
        h16 af[4];
#pragma unroll
        for (int mm = 0; mm < 4; ++mm)
          af[mm] = afrag_ptr(h1 + ((mb + mm) * 16 + l15) * 256, k * 32);
#pragma unroll
        for (int mm = 0; mm < 4; ++mm) {
          c[mm][0] = WMMA_F16(af[mm], bf0, c[mm][0]);
          c[mm][1] = WMMA_F16(af[mm], bf1, c[mm][1]);
        }
      }
#pragma unroll
      for (int mm = 0; mm < 4; ++mm)
#pragma unroll
        for (int j = 0; j < 2; ++j) {
          const int col = wave * 32 + j * 16 + l15;
          const float bi = j ? bi1 : bi0;
#pragma unroll
          for (int v = 0; v < 8; ++v) {
            float val = c[mm][j][v] + bi;
            h2[((mb + mm) * 16 + v + hir) * 256 + col] =
                (_Float16)(val > 0.0f ? val : 0.0f);
          }
        }
    }
  }
  __syncthreads();

  // ---- L3: [128][256] @ ow3t -> out f32, + ob3 (no relu) ----
  {
    f8 c[4];
#pragma unroll
    for (int n = 0; n < 4; ++n)
#pragma unroll
      for (int v = 0; v < 8; ++v) c[n][v] = 0.0f;
#pragma unroll
    for (int k = 0; k < 8; ++k) {
      h16 af = afrag_ptr(h2 + (wave * 16 + l15) * 256, k * 32);
#pragma unroll
      for (int n = 0; n < 4; ++n) {
        h16 bf = *(const h16*)(ow3t + (n * 16 + l15) * 256 + k * 32 + hib);
        c[n] = WMMA_F16(af, bf, c[n]);
      }
    }
#pragma unroll
    for (int n = 0; n < 4; ++n) {
      const float bi = ob3[n * 16 + l15];
#pragma unroll
      for (int v = 0; v < 8; ++v) {
        const int rowg = row0 + wave * 16 + v + hir;
        out[(size_t)rowg * 64 + n * 16 + l15] = c[n][v] + bi;
      }
    }
  }
}

// ---------------------------------------------------------------------------
extern "C" void kernel_launch(void* const* d_in, const int* in_sizes, int n_in,
                              void* d_out, int out_size, void* d_ws, size_t ws_size,
                              hipStream_t stream) {
  const float* x        = (const float*)d_in[0];
  const float* rel_type = (const float*)d_in[1];
  // d_in[2]=rel_rec, d_in[3]=rel_send: structure is known analytically, unused.
  const float* w1  = (const float*)d_in[4];
  const float* b1  = (const float*)d_in[5];
  const float* w2  = (const float*)d_in[6];
  const float* b2  = (const float*)d_in[7];
  const float* ow1 = (const float*)d_in[8];
  const float* ob1 = (const float*)d_in[9];
  const float* ow2 = (const float*)d_in[10];
  const float* ob2 = (const float*)d_in[11];
  const float* ow3 = (const float*)d_in[12];
  const float* ob3 = (const float*)d_in[13];

  char* ws = (char*)d_ws;
  float*    agg  = (float*)(ws + OFF_AGG);
  _Float16* x16  = (_Float16*)(ws + OFF_X16);
  _Float16* w1t  = (_Float16*)(ws + OFF_W1T);
  _Float16* w2t  = (_Float16*)(ws + OFF_W2T);
  _Float16* ow1t = (_Float16*)(ws + OFF_OW1T);
  _Float16* ow2t = (_Float16*)(ws + OFF_OW2T);
  _Float16* ow3t = (_Float16*)(ws + OFF_OW3T);

  prep_kernel<<<1024, 256, 0, stream>>>(x, w1, w2, ow1, ow2, ow3,
                                        x16, w1t, w2t, ow1t, ow2t, ow3t);
  edge_kernel<<<dim3(4, 64), 256, 0, stream>>>(x16, rel_type, w1t, b1, w2t, b2, agg);
  node_kernel<<<32, 256, 0, stream>>>(x16, agg, ow1t, ob1, ow2t, ob2, ow3t, ob3,
                                      (float*)d_out);
}
